// Transformer_60172491816993
// MI455X (gfx1250) — matrix operand order repaired and checked by reading, not compiled, not run on hardware
//
#include <hip/hip_runtime.h>

// ---------------------------------------------------------------------------
// CDNA5 / gfx1250 transformer forward.  All GEMMs + attention use
// V_WMMA_F32_16X16X32_BF16 (fp32 accumulate); double-buffered LDS ping-pong
// (1 barrier / K-step); async global->LDS staging on the ASYNCcnt path.
// ---------------------------------------------------------------------------

typedef __attribute__((ext_vector_type(16))) __bf16 v16bf;
typedef __attribute__((ext_vector_type(8)))  __bf16 v8bf;
typedef __attribute__((ext_vector_type(8)))  float  v8f;
typedef __attribute__((ext_vector_type(4)))  int    v4i;

#define BB   2
#define SS   1024
#define HH   8
#define DD   512
#define DKK  64
#define DFFN 1024
#define LL   6
#define VTGT 32000
#define NEGV (-1.0e9f)

// Async global->LDS staging (gfx1250 ASYNCcnt path); prototype probe-learned:
// both pointer args are v4i* (128-bit chunks).
#if defined(__has_builtin)
#  if __has_builtin(__builtin_amdgcn_global_load_async_to_lds_b128) && \
      __has_builtin(__builtin_amdgcn_s_wait_asynccnt)
#    define ASYNC_LDS 1
#  endif
#endif
#ifndef ASYNC_LDS
#  define ASYNC_LDS 0
#endif

static __device__ __forceinline__ __bf16 f2bf(float f) {
  return (__bf16)f;   // native RNE convert on gfx1250
}

static __device__ __forceinline__ v16bf cat16(v8bf lo, v8bf hi) {
  return __builtin_shufflevector(lo, hi, 0,1,2,3,4,5,6,7,8,9,10,11,12,13,14,15);
}

// ---------------------------------------------------------------------------
// Batched GEMM:  C[z] = A[z] (MxK, f32) * B[z] (KxN, f32) + bias[z]
// flags: 1 = ReLU, 2 = store bf16 (else f32).  Strides in elements.
// Tile: BM=128, BN_ in {64,128}, BK=32; 256 threads = 8 waves.
// Double-buffered LDS: one barrier per K-step, next tile's global loads
// issued before the current tile's WMMAs.
// ---------------------------------------------------------------------------
#define GBM 128
#define GBK 32
#define KP  40   // padded LDS k-stride (bf16 elems); 80B keeps 16B align + bank spread

template <int BN_>
__global__ __launch_bounds__(256) void gemm_bf16_kernel(
    const float* __restrict__ A, int lda, long sAz,
    const float* __restrict__ B, int ldb, long sBz,
    const float* __restrict__ bias, long sBiasz,
    void* __restrict__ C, int ldc, long sCz,
    int M, int N, int K, int flags)
{
  constexpr int NT = BN_ / 32;   // 16-col tiles per wave col-group; also B-stage iters
  constexpr int CG = BN_ / 2;    // col-group width per wave

  __shared__ __align__(16) __bf16 As[2][GBM * KP];
  __shared__ __align__(16) __bf16 Bs[2][BN_ * KP];   // stored transposed: [n][k]

  const int z = blockIdx.z;
  A += (long)z * sAz;
  B += (long)z * sBz;
  const float* bz = bias ? (bias + (long)z * sBiasz) : nullptr;

  const int m0 = blockIdx.y * GBM;
  const int n0 = blockIdx.x * BN_;
  const int t    = threadIdx.x;
  const int lane = t & 31;
  const int wv   = t >> 5;
  const int wr   = wv & 3, wc = wv >> 2;
  const int half = lane >> 4, l16 = lane & 15;

  v8f acc[2][NT] = {};
  float4 aT[4], bT[NT];

  auto loadTile = [&](int k0) {
    #pragma unroll
    for (int i = 0; i < 4; ++i) {
      int idx = t + i * 256;
      int r = idx >> 3, c4 = (idx & 7) << 2;
      aT[i] = *(const float4*)(A + (long)(m0 + r) * lda + k0 + c4);
    }
    #pragma unroll
    for (int i = 0; i < NT; ++i) {
      int idx = t + i * 256;
      int kk = idx / (BN_ / 4), c4 = (idx % (BN_ / 4)) << 2;
      bT[i] = *(const float4*)(B + (long)(k0 + kk) * ldb + n0 + c4);
    }
  };
  auto storeTile = [&](int buf) {
    #pragma unroll
    for (int i = 0; i < 4; ++i) {
      int idx = t + i * 256;
      int r = idx >> 3, c4 = (idx & 7) << 2;
      __bf16* d = &As[buf][r * KP + c4];
      d[0] = f2bf(aT[i].x); d[1] = f2bf(aT[i].y); d[2] = f2bf(aT[i].z); d[3] = f2bf(aT[i].w);
    }
    #pragma unroll
    for (int i = 0; i < NT; ++i) {
      int idx = t + i * 256;
      int kk = idx / (BN_ / 4), c4 = (idx % (BN_ / 4)) << 2;
      Bs[buf][(c4 + 0) * KP + kk] = f2bf(bT[i].x);
      Bs[buf][(c4 + 1) * KP + kk] = f2bf(bT[i].y);
      Bs[buf][(c4 + 2) * KP + kk] = f2bf(bT[i].z);
      Bs[buf][(c4 + 3) * KP + kk] = f2bf(bT[i].w);
    }
  };

  // prologue: tile 0 into buffer 0
  loadTile(0);
  storeTile(0);
  __syncthreads();

  int cur = 0;
  for (int k0 = 0; k0 < K; k0 += GBK) {
    const bool hasNext = (k0 + GBK) < K;
    if (hasNext) {
      loadTile(k0 + GBK);           // global loads in flight during WMMAs
      if (k0 + 2 * GBK < K) {       // global_prefetch_b8 of tile after next
        __builtin_prefetch(A + (long)(m0 + (t >> 1)) * lda + k0 + 2 * GBK + (t & 1) * 16, 0, 0);
        __builtin_prefetch(B + (long)(k0 + 2 * GBK + (t >> 3)) * ldb + n0 + (t & 7) * (BN_ / 8), 0, 0);
      }
    }

    // A fragment (16x32 bf16): lane row = l16, chunks at k=half*8 and 16+half*8
    v16bf af[2], bfv[NT];
    #pragma unroll
    for (int i = 0; i < 2; ++i) {
      const __bf16* p = &As[cur][(wr * 32 + i * 16 + l16) * KP + half * 8];
      af[i] = cat16(*(const v8bf*)p, *(const v8bf*)(p + 16));
    }
    // B fragment (32x16 bf16): lane col = l16, chunks at k=half*8 and 16+half*8
    #pragma unroll
    for (int j = 0; j < NT; ++j) {
      const __bf16* p = &Bs[cur][(wc * CG + j * 16 + l16) * KP + half * 8];
      bfv[j] = cat16(*(const v8bf*)p, *(const v8bf*)(p + 16));
    }
    #pragma unroll
    for (int i = 0; i < 2; ++i)
      #pragma unroll
      for (int j = 0; j < NT; ++j)
        acc[i][j] = __builtin_amdgcn_wmma_f32_16x16x32_bf16(
            false, af[i], false, bfv[j], (short)0, acc[i][j], false, false);

    if (hasNext) storeTile(cur ^ 1);
    __syncthreads();
    cur ^= 1;
  }

  // epilogue: C layout — lane holds col l16, rows half*8 + r (vgpr r)
  #pragma unroll
  for (int i = 0; i < 2; ++i) {
    #pragma unroll
    for (int j = 0; j < NT; ++j) {
      int n = n0 + wc * CG + j * 16 + l16;
      float bvv = bz ? bz[n] : 0.0f;
      #pragma unroll
      for (int r = 0; r < 8; ++r) {
        int m = m0 + wr * 32 + i * 16 + half * 8 + r;
        float v = acc[i][j][r] + bvv;
        if (flags & 1) v = fmaxf(v, 0.0f);
        if (flags & 2) {
          __bf16* Cb = (__bf16*)C + (long)z * sCz;
          Cb[(long)m * ldc + n] = f2bf(v);
        } else {
          float* Cf = (float*)C + (long)z * sCz;
          Cf[(long)m * ldc + n] = v;
        }
      }
    }
  }
}

// ---------------------------------------------------------------------------
// Flash attention.  Q,K,V bf16 in [b*S + s][H*64] layout (head h at col h*64).
// Block = (q-block of 128, b*H).  8 waves, each wave owns 16 q rows.
// Streaming softmax; scores and PV through WMMA bf16.  K/V double-buffered:
// next key-block staged (async K copy + V transpose) during compute, one
// barrier per key-block.  P->PV is same-wave LDS (in-order, no barrier).
// ---------------------------------------------------------------------------
__global__ __launch_bounds__(256) void attn_kernel(
    const __bf16* __restrict__ Q, const __bf16* __restrict__ K,
    const __bf16* __restrict__ V, const int* __restrict__ padseq,
    int causal, float* __restrict__ O)
{
  __shared__ __align__(16) __bf16 Qs[128 * 72];        // [row][dk]
  __shared__ __align__(16) __bf16 Ks[2][32 * 72];      // [key][dk]
  __shared__ __align__(16) __bf16 Vt[2][64 * 40];      // [dv][key] (transposed)
  __shared__ __align__(16) __bf16 Ps[8 * 16 * 40];     // per-wave P staging

  const int bh = blockIdx.y;
  const int b = bh / HH, h = bh % HH;
  const int qb = blockIdx.x * 128;
  const int t = threadIdx.x;
  const int lane = t & 31, w = t >> 5;
  const int half = lane >> 4, l16 = lane & 15;
  const float scale = 0.125f;   // 1/sqrt(64)

  auto stageKV = [&](int kb, int buf) {
    int key = t >> 3, c8 = (t & 7) << 3;
    const __bf16* gk = K + (long)(b * SS + kb + key) * (HH * DKK) + h * DKK + c8;
#if ASYNC_LDS
    __builtin_amdgcn_global_load_async_to_lds_b128(
        (v4i*)gk, (v4i*)&Ks[buf][key * 72 + c8], 0, 0);
#else
    *(v8bf*)&Ks[buf][key * 72 + c8] = *(const v8bf*)gk;
#endif
    v8bf vv = *(const v8bf*)(V + (long)(b * SS + kb + key) * (HH * DKK) + h * DKK + c8);
    #pragma unroll
    for (int e = 0; e < 8; ++e) Vt[buf][(c8 + e) * 40 + key] = vv[e];
  };

  // ---- stage Q block (128 x 64 bf16) + first K/V block ----
#if ASYNC_LDS
  #pragma unroll
  for (int i = 0; i < 4; ++i) {
    int idx = t + i * 256;
    int row = idx >> 3, c8 = (idx & 7) << 3;
    const __bf16* g = Q + (long)(b * SS + qb + row) * (HH * DKK) + h * DKK + c8;
    __builtin_amdgcn_global_load_async_to_lds_b128(
        (v4i*)g, (v4i*)(Qs + row * 72 + c8), 0, 0);
  }
#else
  {
    v8bf qT[4];
    #pragma unroll
    for (int i = 0; i < 4; ++i) {
      int idx = t + i * 256;
      int row = idx >> 3, c8 = (idx & 7) << 3;
      qT[i] = *(const v8bf*)(Q + (long)(b * SS + qb + row) * (HH * DKK) + h * DKK + c8);
    }
    #pragma unroll
    for (int i = 0; i < 4; ++i) {
      int idx = t + i * 256;
      int row = idx >> 3, c8 = (idx & 7) << 3;
      *(v8bf*)(Qs + row * 72 + c8) = qT[i];
    }
  }
#endif
  stageKV(0, 0);
#if ASYNC_LDS
  __builtin_amdgcn_s_wait_asynccnt(0);
#endif
  __syncthreads();

  v16bf qa[2];
  #pragma unroll
  for (int d = 0; d < 2; ++d) {
    const __bf16* p = Qs + (w * 16 + l16) * 72 + d * 32 + half * 8;
    qa[d] = cat16(*(const v8bf*)p, *(const v8bf*)(p + 16));
  }

  float mrow[8], lrow[8], alpha[8];
  v8f o[4] = {};
  #pragma unroll
  for (int r = 0; r < 8; ++r) { mrow[r] = -3.0e38f; lrow[r] = 0.0f; }

  int cur = 0;
  for (int kb = 0; kb < SS; kb += 32) {
    if (causal && kb > qb + 127) break;   // uniform across block

    const int nxt = kb + 32;
    if (nxt < SS) stageKV(nxt, cur ^ 1);  // overlapped with the whole compute

    // scores: two 16-key sub-tiles, k-dim = 64 dk via 2 WMMAs each
    v8f s[2] = {};
    #pragma unroll
    for (int sub = 0; sub < 2; ++sub) {
      #pragma unroll
      for (int d = 0; d < 2; ++d) {
        const __bf16* p = &Ks[cur][(sub * 16 + l16) * 72 + d * 32 + half * 8];
        v16bf kf = cat16(*(const v8bf*)p, *(const v8bf*)(p + 16));
        s[sub] = __builtin_amdgcn_wmma_f32_16x16x32_bf16(
            false, qa[d], false, kf, (short)0, s[sub], false, false);
      }
    }

    // scale + mask (pad keys, causal)
    #pragma unroll
    for (int sub = 0; sub < 2; ++sub) {
      int key = kb + sub * 16 + l16;
      bool pad = (padseq[b * SS + key] == 0);
      #pragma unroll
      for (int r = 0; r < 8; ++r) {
        int q = qb + w * 16 + half * 8 + r;
        float v = s[sub][r] * scale;
        if (pad || (causal && key > q)) v = NEGV;
        s[sub][r] = v;
      }
    }

    // streaming softmax: row reductions across the 16-lane half
    #pragma unroll
    for (int r = 0; r < 8; ++r) {
      float mx = fmaxf(s[0][r], s[1][r]);
      for (int off = 1; off < 16; off <<= 1) mx = fmaxf(mx, __shfl_xor(mx, off, 32));
      float mnew = fmaxf(mrow[r], mx);
      float a = __expf(mrow[r] - mnew);
      float p0 = __expf(s[0][r] - mnew);
      float p1 = __expf(s[1][r] - mnew);
      s[0][r] = p0; s[1][r] = p1;
      float sum = p0 + p1;
      for (int off = 1; off < 16; off <<= 1) sum += __shfl_xor(sum, off, 32);
      lrow[r] = lrow[r] * a + sum;
      mrow[r] = mnew;
      alpha[r] = a;
    }
    #pragma unroll
    for (int t4 = 0; t4 < 4; ++t4)
      #pragma unroll
      for (int r = 0; r < 8; ++r) o[t4][r] *= alpha[r];

    // stage P (C layout -> A layout) through per-wave LDS; same-wave LDS ops
    // are processed in order, so no block barrier is needed before the reads.
    __bf16* Pw = Ps + w * 640;
    #pragma unroll
    for (int sub = 0; sub < 2; ++sub)
      #pragma unroll
      for (int r = 0; r < 8; ++r)
        Pw[(half * 8 + r) * 40 + sub * 16 + l16] = f2bf(s[sub][r]);

    { // O += P (16x32) * V (32x64)
      const __bf16* p = Pw + l16 * 40 + half * 8;
      v16bf pa = cat16(*(const v8bf*)p, *(const v8bf*)(p + 16));
      #pragma unroll
      for (int t4 = 0; t4 < 4; ++t4) {
        const __bf16* vp = &Vt[cur][(t4 * 16 + l16) * 40 + half * 8];
        v16bf vf = cat16(*(const v8bf*)vp, *(const v8bf*)(vp + 16));
        o[t4] = __builtin_amdgcn_wmma_f32_16x16x32_bf16(
            false, pa, false, vf, (short)0, o[t4], false, false);
      }
    }

#if ASYNC_LDS
    __builtin_amdgcn_s_wait_asynccnt(0);   // own async K copy done
#endif
    __syncthreads();                        // next buffer visible; cur reusable
    cur ^= 1;
  }

  // normalize + store fp32, heads concatenated along columns
  #pragma unroll
  for (int r = 0; r < 8; ++r) {
    float inv = lrow[r] > 0.0f ? 1.0f / lrow[r] : 0.0f;
    int q = qb + w * 16 + half * 8 + r;
    #pragma unroll
    for (int t4 = 0; t4 < 4; ++t4)
      O[(long)(b * SS + q) * (HH * DKK) + h * DKK + t4 * 16 + l16] = o[t4][r] * inv;
  }
}

// ---------------------------------------------------------------------------
// x = LayerNorm(a + x) * g + b     (one block per row, D = 512, 256 threads)
// ---------------------------------------------------------------------------
__global__ __launch_bounds__(256) void add_ln_kernel(
    const float* __restrict__ a, float* __restrict__ x,
    const float* __restrict__ g, const float* __restrict__ b, int D)
{
  __shared__ float red[256];
  const long base = (long)blockIdx.x * D;
  const int t = threadIdx.x;
  float v0 = a[base + t] + x[base + t];
  float v1 = a[base + t + 256] + x[base + t + 256];
  red[t] = v0 + v1;
  __syncthreads();
  for (int off = 128; off > 0; off >>= 1) {
    if (t < off) red[t] += red[t + off];
    __syncthreads();
  }
  float mu = red[0] / (float)D;
  __syncthreads();
  float d0 = v0 - mu, d1 = v1 - mu;
  red[t] = d0 * d0 + d1 * d1;
  __syncthreads();
  for (int off = 128; off > 0; off >>= 1) {
    if (t < off) red[t] += red[t + off];
    __syncthreads();
  }
  float rs = rsqrtf(red[0] / (float)D + 1e-5f);
  x[base + t]       = d0 * rs * g[t]       + b[t];
  x[base + t + 256] = d1 * rs * g[t + 256] + b[t + 256];
}

// ---------------------------------------------------------------------------
// out[bs, d] = emb[seq[bs], d] + sinusoid PE(pos[bs], d)
// ---------------------------------------------------------------------------
__global__ __launch_bounds__(256) void embed_kernel(
    const int* __restrict__ seq, const int* __restrict__ pos,
    const float* __restrict__ emb, float* __restrict__ out, int BSn, int D)
{
  long idx = (long)blockIdx.x * blockDim.x + threadIdx.x;
  int bs = (int)(idx / D);
  int d  = (int)(idx % D);
  if (bs >= BSn) return;
  int tok = seq[bs];
  int p   = pos[bs];
  float freq = __expf(-(float)(d & ~1) * (9.210340372f / (float)D));  // ln(1e4)
  float ang  = (float)p * freq;
  float pe   = (d & 1) ? __cosf(ang) : __sinf(ang);
  out[idx] = emb[(long)tok * D + d] + pe;
}

// ---------------------------------------------------------------------------
extern "C" void kernel_launch(void* const* d_in, const int* in_sizes, int n_in,
                              void* d_out, int out_size, void* d_ws, size_t ws_size,
                              hipStream_t stream)
{
  (void)in_sizes; (void)n_in; (void)out_size; (void)ws_size;

  const int*   src_seq        = (const int*)d_in[0];
  const int*   src_pos        = (const int*)d_in[1];
  const int*   tgt_seq        = (const int*)d_in[2];
  const int*   tgt_pos        = (const int*)d_in[3];
  const float* src_emb        = (const float*)d_in[4];
  const float* tgt_emb        = (const float*)d_in[5];
  const float* enc_attn_w     = (const float*)d_in[6];
  const float* enc_attn_b     = (const float*)d_in[7];
  const float* enc_out_w      = (const float*)d_in[8];
  const float* enc_out_b      = (const float*)d_in[9];
  const float* enc_ln         = (const float*)d_in[10];
  const float* dec_self_w     = (const float*)d_in[11];
  const float* dec_self_b     = (const float*)d_in[12];
  const float* dec_self_out_w = (const float*)d_in[13];
  const float* dec_self_out_b = (const float*)d_in[14];
  const float* dec_cross_w    = (const float*)d_in[15];
  const float* dec_cross_b    = (const float*)d_in[16];
  const float* dec_cross_out_w= (const float*)d_in[17];
  const float* dec_cross_out_b= (const float*)d_in[18];
  const float* dec_ffn_w1     = (const float*)d_in[19];
  const float* dec_ffn_b1     = (const float*)d_in[20];
  const float* dec_ffn_w2     = (const float*)d_in[21];
  const float* dec_ffn_b2     = (const float*)d_in[22];
  const float* dec_ln         = (const float*)d_in[23];
  const float* proj_w         = (const float*)d_in[24];

  // workspace layout (~30 MB)
  char* ws = (char*)d_ws;
  const size_t MBy = 1024 * 1024;
  float*  xe   = (float*)(ws + 0 * MBy);    // encoder state / enc_out (4 MB)
  float*  y    = (float*)(ws + 4 * MBy);    // decoder state (4 MB)
  __bf16* qbuf = (__bf16*)(ws + 8 * MBy);   // bf16 Q (2 MB)
  __bf16* kbuf = (__bf16*)(ws + 10 * MBy);  // bf16 K (2 MB)
  __bf16* vbuf = (__bf16*)(ws + 12 * MBy);  // bf16 V (2 MB)
  float*  obuf = (float*)(ws + 14 * MBy);   // attention output (4 MB)
  float*  tmp  = (float*)(ws + 18 * MBy);   // projection result (4 MB)
  float*  hid  = (float*)(ws + 22 * MBy);   // FFN hidden (8 MB)

  const int M = BB * SS;   // 2048

  auto gemm = [&](const float* A, int lda, long sAz,
                  const float* Bm, int ldb, long sBz,
                  const float* bias, long sBiasz,
                  void* C, int ldc, long sCz,
                  int m, int n, int k, int Z, int flags) {
    if (n % 128 == 0) {
      dim3 grid(n / 128, m / GBM, Z);
      gemm_bf16_kernel<128><<<grid, 256, 0, stream>>>(A, lda, sAz, Bm, ldb, sBz,
                                                      bias, sBiasz, C, ldc, sCz,
                                                      m, n, k, flags);
    } else {
      dim3 grid(n / 64, m / GBM, Z);
      gemm_bf16_kernel<64><<<grid, 256, 0, stream>>>(A, lda, sAz, Bm, ldb, sBz,
                                                     bias, sBiasz, C, ldc, sCz,
                                                     m, n, k, flags);
    }
  };

  const long wHead = (long)DD * DKK;           // per-head weight stride
  const long wMat  = (long)HH * DD * DKK;      // per-matrix (q/k/v) stride
  const long bMat  = (long)HH * DKK;

  // ======================= encoder =======================
  embed_kernel<<<(M * DD) / 256, 256, 0, stream>>>(src_seq, src_pos, src_emb, xe, M, DD);
  for (int i = 0; i < LL; ++i) {
    const float* w  = enc_attn_w + (size_t)i * 3 * wMat;
    const float* bi = enc_attn_b + (size_t)i * 3 * bMat;
    gemm(xe, DD, 0, w + 0 * wMat, DKK, wHead, bi + 0 * bMat, DKK, qbuf, DD, 64, M, DKK, DD, HH, 2);
    gemm(xe, DD, 0, w + 1 * wMat, DKK, wHead, bi + 1 * bMat, DKK, kbuf, DD, 64, M, DKK, DD, HH, 2);
    gemm(xe, DD, 0, w + 2 * wMat, DKK, wHead, bi + 2 * bMat, DKK, vbuf, DD, 64, M, DKK, DD, HH, 2);
    attn_kernel<<<dim3(SS / 128, BB * HH), 256, 0, stream>>>(qbuf, kbuf, vbuf, src_seq, 0, obuf);
    gemm(obuf, DD, 0, enc_out_w + (size_t)i * DD * DD, DD, 0,
         enc_out_b + (size_t)i * DD, 0, tmp, DD, 0, M, DD, DD, 1, 0);
    add_ln_kernel<<<M, 256, 0, stream>>>(tmp, xe,
        enc_ln + (size_t)(i * 2 + 0) * DD, enc_ln + (size_t)(i * 2 + 1) * DD, DD);
  }

  // ======================= decoder =======================
  embed_kernel<<<(M * DD) / 256, 256, 0, stream>>>(tgt_seq, tgt_pos, tgt_emb, y, M, DD);
  for (int i = 0; i < LL; ++i) {
    // --- masked self-attention ---
    const float* w  = dec_self_w + (size_t)i * 3 * wMat;
    const float* bi = dec_self_b + (size_t)i * 3 * bMat;
    gemm(y, DD, 0, w + 0 * wMat, DKK, wHead, bi + 0 * bMat, DKK, qbuf, DD, 64, M, DKK, DD, HH, 2);
    gemm(y, DD, 0, w + 1 * wMat, DKK, wHead, bi + 1 * bMat, DKK, kbuf, DD, 64, M, DKK, DD, HH, 2);
    gemm(y, DD, 0, w + 2 * wMat, DKK, wHead, bi + 2 * bMat, DKK, vbuf, DD, 64, M, DKK, DD, HH, 2);
    attn_kernel<<<dim3(SS / 128, BB * HH), 256, 0, stream>>>(qbuf, kbuf, vbuf, tgt_seq, 1, obuf);
    gemm(obuf, DD, 0, dec_self_out_w + (size_t)i * DD * DD, DD, 0,
         dec_self_out_b + (size_t)i * DD, 0, tmp, DD, 0, M, DD, DD, 1, 0);
    add_ln_kernel<<<M, 256, 0, stream>>>(tmp, y,
        dec_ln + ((size_t)(i * 3 + 0) * 2 + 0) * DD,
        dec_ln + ((size_t)(i * 3 + 0) * 2 + 1) * DD, DD);

    // --- cross-attention (K,V from encoder output) ---
    const float* wx = dec_cross_w + (size_t)i * 3 * wMat;
    const float* bx = dec_cross_b + (size_t)i * 3 * bMat;
    gemm(y,  DD, 0, wx + 0 * wMat, DKK, wHead, bx + 0 * bMat, DKK, qbuf, DD, 64, M, DKK, DD, HH, 2);
    gemm(xe, DD, 0, wx + 1 * wMat, DKK, wHead, bx + 1 * bMat, DKK, kbuf, DD, 64, M, DKK, DD, HH, 2);
    gemm(xe, DD, 0, wx + 2 * wMat, DKK, wHead, bx + 2 * bMat, DKK, vbuf, DD, 64, M, DKK, DD, HH, 2);
    attn_kernel<<<dim3(SS / 128, BB * HH), 256, 0, stream>>>(qbuf, kbuf, vbuf, src_seq, 0, obuf);
    gemm(obuf, DD, 0, dec_cross_out_w + (size_t)i * DD * DD, DD, 0,
         dec_cross_out_b + (size_t)i * DD, 0, tmp, DD, 0, M, DD, DD, 1, 0);
    add_ln_kernel<<<M, 256, 0, stream>>>(tmp, y,
        dec_ln + ((size_t)(i * 3 + 1) * 2 + 0) * DD,
        dec_ln + ((size_t)(i * 3 + 1) * 2 + 1) * DD, DD);

    // --- FFN ---
    gemm(y, DD, 0, dec_ffn_w1 + (size_t)i * DD * DFFN, DFFN, 0,
         dec_ffn_b1 + (size_t)i * DFFN, 0, hid, DFFN, 0, M, DFFN, DD, 1, 1 /*relu*/);
    gemm(hid, DFFN, 0, dec_ffn_w2 + (size_t)i * DFFN * DD, DD, 0,
         dec_ffn_b2 + (size_t)i * DD, 0, tmp, DD, 0, M, DD, DFFN, 1, 0);
    add_ln_kernel<<<M, 256, 0, stream>>>(tmp, y,
        dec_ln + ((size_t)(i * 3 + 2) * 2 + 0) * DD,
        dec_ln + ((size_t)(i * 3 + 2) * 2 + 1) * DD, DD);
  }

  // ======================= final projection =======================
  gemm(y, DD, 0, proj_w, VTGT, 0, nullptr, 0, (float*)d_out, VTGT, 0, M, VTGT, DD, 1, 0);
}
